// BiLSTM_Proto_77988016161347
// MI455X (gfx1250) — matrix operand order repaired and checked by reading, not compiled, not run on hardware
//
#include <hip/hip_runtime.h>

// ---------------------------------------------------------------------------
// Types for CDNA5 WMMA (wave32)
// ---------------------------------------------------------------------------
typedef __attribute__((ext_vector_type(16))) __bf16 v16bf;
typedef __attribute__((ext_vector_type(8)))  __bf16 v8bf;
typedef __attribute__((ext_vector_type(8)))  float  v8f;
typedef __attribute__((ext_vector_type(4)))  int    v4i_;

// Optional CDNA5 async global->LDS path (compile-safe fallback otherwise)
#if defined(__has_builtin)
#  if __has_builtin(__builtin_amdgcn_global_load_async_to_lds_b128) && \
      __has_builtin(__builtin_amdgcn_s_wait_asynccnt)
#    define USE_ASYNC_LDS 1
#  endif
#endif
#ifndef USE_ASYNC_LDS
#  define USE_ASYNC_LDS 0
#endif

#if USE_ASYNC_LDS
// toolchain signature (from diagnostics): (int4 AS1*, int4 AS3*, int, int)
#define ASYNC_G2L_B128(gsrc, ldst)                                         \
  __builtin_amdgcn_global_load_async_to_lds_b128(                          \
      (__attribute__((address_space(1))) v4i_*)(gsrc),                     \
      (__attribute__((address_space(3))) v4i_*)(ldst), 0, 0)
#endif

// Problem constants
constexpr int B_  = 32;
constexpr int T_  = 512;
constexpr int E_  = 512;
constexpr int H2_ = 512;    // per-direction hidden
constexpr int G4_ = 2048;   // 4*H2
constexpr int HD_ = 1024;   // concat hidden
constexpr int P_  = 128;    // prototypes
constexpr int ROWS_ = B_ * T_;        // 16384
constexpr int XGSTEP_ = B_ * G4_;     // floats per (dir,t) block of xg = 65536

// Workspace layout (bytes, all 256-aligned)
constexpr size_t OFF_XBF  = 0;                              // x bf16 [16384][512]
constexpr size_t OFF_WIH  = OFF_XBF  + (size_t)ROWS_*E_*2;  // w_ih bf16 [2][2048][512]
constexpr size_t OFF_WHH  = OFF_WIH  + (size_t)2*G4_*E_*2;  // w_hh bf16 [2][2048][512]
constexpr size_t OFF_PRO  = OFF_WHH  + (size_t)2*G4_*H2_*2; // protos bf16 [128][1024]
constexpr size_t OFF_BIAS = OFF_PRO  + (size_t)P_*HD_*2;    // bias f32 [2][2048]
constexpr size_t OFF_P2   = OFF_BIAS + (size_t)2*G4_*4;     // p2 f32 [128] (padded)
constexpr size_t OFF_XG   = OFF_P2   + 1024;                // xg f32 [2][512][65536] frag-major
constexpr size_t OFF_HBUF = OFF_XG   + (size_t)2*T_*B_*G4_*4; // h bf16 [2][2][32][512]
constexpr size_t OFF_LSTM = OFF_HBUF + (size_t)2*2*B_*H2_*2;  // lstm_out bf16 [16384][1024]
constexpr size_t OFF_X2   = OFF_LSTM + (size_t)ROWS_*HD_*2;   // x2 f32 [16384]
constexpr size_t OFF_BAR  = OFF_X2   + (size_t)ROWS_*4;       // barriers (uints)

// ---------------------------------------------------------------------------
// WMMA helpers
// ---------------------------------------------------------------------------
__device__ __forceinline__ v8f wmma_bf16(v16bf a, v16bf b, v8f c) {
  return __builtin_amdgcn_wmma_f32_16x16x32_bf16(false, a, false, b,
                                                 (short)0, c, false, false);
}

// A-matrix 16x32 bf16 fragment. base points at row 0 of the 16-row tile.
// Lane lr (=lane&15) supplies row lr; sel (=lane>>4) picks K {sel*16..+15}.
__device__ __forceinline__ v16bf load_a_tile(const __bf16* __restrict__ base,
                                             int stride, int lr, int sel, int k0) {
  const __bf16* p = base + (size_t)lr * stride + k0;
  v8bf lo = *reinterpret_cast<const v8bf*>(p + sel * 16);
  v8bf hi = *reinterpret_cast<const v8bf*>(p + sel * 16 + 8);
  v16bf f;
#pragma unroll
  for (int i = 0; i < 8; ++i) { f[i] = lo[i]; f[i + 8] = hi[i]; }
  return f;
}

// B-matrix 32x16 bf16 fragment from global: lane supplies column `col`,
// 16 contiguous K values starting at k0 + sel*16.
__device__ __forceinline__ v16bf load_b_tile(const __bf16* __restrict__ w,
                                             int col, int stride, int sel, int k0) {
  const __bf16* p = w + (size_t)col * stride + k0 + sel * 16;
  v8bf lo = *reinterpret_cast<const v8bf*>(p);
  v8bf hi = *reinterpret_cast<const v8bf*>(p + 8);
  v16bf f;
#pragma unroll
  for (int i = 0; i < 8; ++i) { f[i] = lo[i]; f[i + 8] = hi[i]; }
  return f;
}

__device__ __forceinline__ float sigmoidf_(float x) {
  return 1.0f / (1.0f + __expf(-x));
}

// Device-wide split barrier over `nblocks` blocks (one counter+gen pair).
__device__ __forceinline__ void grid_barrier(unsigned* counter, unsigned* gen,
                                             unsigned nblocks) {
  __syncthreads();
  if (threadIdx.x == 0) {
    __threadfence();
    unsigned g = __hip_atomic_load(gen, __ATOMIC_ACQUIRE, __HIP_MEMORY_SCOPE_AGENT);
    unsigned arrived = __hip_atomic_fetch_add(counter, 1u, __ATOMIC_ACQ_REL,
                                              __HIP_MEMORY_SCOPE_AGENT);
    if (arrived == nblocks - 1u) {
      __hip_atomic_store(counter, 0u, __ATOMIC_RELAXED, __HIP_MEMORY_SCOPE_AGENT);
      __hip_atomic_fetch_add(gen, 1u, __ATOMIC_ACQ_REL, __HIP_MEMORY_SCOPE_AGENT);
    } else {
      while (__hip_atomic_load(gen, __ATOMIC_ACQUIRE, __HIP_MEMORY_SCOPE_AGENT) == g) {
        __builtin_amdgcn_s_sleep(1);
      }
    }
  }
  __syncthreads();
}

// ---------------------------------------------------------------------------
// Prep kernels
// ---------------------------------------------------------------------------
__global__ void k_ws_init(__bf16* hbuf, unsigned* bar) {
  int i = blockIdx.x * blockDim.x + threadIdx.x;
  if (i < 2 * 2 * B_ * H2_) hbuf[i] = (__bf16)0.0f;
  if (i < 8) bar[i] = 0u;
}

__global__ void k_embed_bf(const int* __restrict__ ids,
                           const float* __restrict__ tab,
                           __bf16* __restrict__ xbf) {
  int row = blockIdx.x;                       // b*T + t
  int id  = ids[row];
  const float* s = tab + (size_t)id * E_;
  __bf16* d = xbf + (size_t)row * E_;
  for (int i = threadIdx.x; i < E_; i += blockDim.x) d[i] = (__bf16)s[i];
}

__global__ void k_f2bf(const float* __restrict__ src, __bf16* __restrict__ dst, int n) {
  int i = blockIdx.x * blockDim.x + threadIdx.x;
  if (i < n) dst[i] = (__bf16)src[i];
}

__global__ void k_bias_sum(const float* bi_f, const float* bh_f,
                           const float* bi_b, const float* bh_b, float* bias) {
  int i = blockIdx.x * blockDim.x + threadIdx.x;
  if (i < G4_) {
    bias[i]       = bi_f[i] + bh_f[i];
    bias[G4_ + i] = bi_b[i] + bh_b[i];
  }
}

__global__ void k_p2(const float* __restrict__ pro, float* __restrict__ p2) {
  __shared__ float red[256];
  int row = blockIdx.x;
  const float* p = pro + (size_t)row * HD_;
  float s = 0.0f;
  for (int i = threadIdx.x; i < HD_; i += 256) { float v = p[i]; s += v * v; }
  red[threadIdx.x] = s;
  __syncthreads();
  for (int o = 128; o > 0; o >>= 1) {
    if (threadIdx.x < o) red[threadIdx.x] += red[threadIdx.x + o];
    __syncthreads();
  }
  if (threadIdx.x == 0) p2[row] = red[0];
}

__global__ void k_row_sumsq(const __bf16* __restrict__ lstm, float* __restrict__ x2) {
  __shared__ float red[256];
  int row = blockIdx.x;
  const __bf16* p = lstm + (size_t)row * HD_;
  float s = 0.0f;
  for (int i = threadIdx.x; i < HD_; i += 256) { float v = (float)p[i]; s += v * v; }
  red[threadIdx.x] = s;
  __syncthreads();
  for (int o = 128; o > 0; o >>= 1) {
    if (threadIdx.x < o) red[threadIdx.x] += red[threadIdx.x + o];
    __syncthreads();
  }
  if (threadIdx.x == 0) x2[row] = red[0];
}

// ---------------------------------------------------------------------------
// Input projection: xg = x . w_ihT + bias, stored FRAGMENT-MAJOR:
//   xg[dir][t][ ((b>>4)*4+gate)*32 + ngrp ][lane=dsel*16+(n&15)][r=b&7]
// so the recurrence seeds each accumulator with one contiguous v8f per gate.
// grid = (N/128, M/64, 2), block = 256 (8 waves). Wave tile: 16M x 64N.
// ---------------------------------------------------------------------------
__global__ void k_xg_gemm(const __bf16* __restrict__ xbf,
                          const __bf16* __restrict__ wih,
                          const float* __restrict__ bias,
                          float* __restrict__ xg) {
  const int dir  = blockIdx.z;
  const int wave = threadIdx.x >> 5;
  const int lane = threadIdx.x & 31;
  const int lr = lane & 15, sel = lane >> 4;
  const int msub = wave >> 1, nch = wave & 1;
  const int M0 = blockIdx.y * 64 + msub * 16;
  const int Nb = blockIdx.x * 128 + nch * 64;

  const __bf16* w  = wih + (size_t)dir * G4_ * E_;
  const float*  bs = bias + (size_t)dir * G4_;
  float*        out = xg + (size_t)dir * T_ * XGSTEP_;

  v8f acc[4];
#pragma unroll
  for (int j = 0; j < 4; ++j) {
    float bv = bs[Nb + j * 16 + lr];
#pragma unroll
    for (int r = 0; r < 8; ++r) acc[j][r] = bv;
  }

  const __bf16* abase = xbf + (size_t)M0 * E_;
  for (int k0 = 0; k0 < E_; k0 += 32) {
    v16bf a = load_a_tile(abase, E_, lr, sel, k0);
#pragma unroll
    for (int j = 0; j < 4; ++j) {
      v16bf b = load_b_tile(w, Nb + j * 16 + lr, E_, sel, k0);
      acc[j] = wmma_bf16(a, b, acc[j]);
    }
  }

#pragma unroll
  for (int j = 0; j < 4; ++j) {
    const int n    = Nb + j * 16 + lr;
    const int gate = n >> 9;           // which of i,f,g,o
    const int ngrp = (n >> 4) & 31;    // 16-col group within the gate
#pragma unroll
    for (int r = 0; r < 8; ++r) {
      int mrow = M0 + r + sel * 8;     // global row = b*T + t
      int b = mrow >> 9, t = mrow & (T_ - 1);
      int dsel  = (b >> 3) & 1;
      int tile  = ((b >> 4) * 4 + gate) * 32 + ngrp;
      size_t off = (size_t)t * XGSTEP_ +
                   (size_t)(tile * 32 + dsel * 16 + lr) * 8 + (b & 7);
      out[off] = acc[j][r];
    }
  }
}

// ---------------------------------------------------------------------------
// Persistent bidirectional LSTM recurrence.
// grid = 64 blocks x 64 threads. dir = blockIdx>>5 ; 32 blocks per direction.
// Each wave owns one 16(batch) x 16(hidden) tile and all 4 gates for it.
// W_hh B-fragments for the block (64 KB) are staged once into LDS (async
// global->LDS when available), so the 512-step critical path reads weights
// at LDS latency. Cell state c stays in VGPRs; h is double-buffered in
// global bf16 (32 KB, L2-resident) behind one grid barrier per step.
// ---------------------------------------------------------------------------
__global__ void k_lstm_recurrence(const float* __restrict__ xg,
                                  const __bf16* __restrict__ whh,
                                  __bf16* __restrict__ hbuf,
                                  __bf16* __restrict__ lstm_out,
                                  unsigned* __restrict__ bar) {
  // [gate][kchunk][lane][16 bf16] = 4*16*32*16 bf16 = 64 KB
  __shared__ __bf16 wsm[4 * 16 * 32 * 16];

  const int dir  = blockIdx.x >> 5;
  const int blk  = blockIdx.x & 31;
  const int wave = threadIdx.x >> 5;   // 0,1 -> batch halves
  const int lane = threadIdx.x & 31;
  const int lr = lane & 15, sel = lane >> 4;
  const int m0 = wave * 16;            // batch rows [m0, m0+16)
  const int n0 = blk * 16;             // hidden cols [n0, n0+16)

  const float*  xg_d  = xg  + (size_t)dir * T_ * XGSTEP_;
  const __bf16* whh_d = whh + (size_t)dir * G4_ * H2_;
  __bf16*       hb    = hbuf + (size_t)dir * 2 * B_ * H2_;
  unsigned*     cnt   = bar + dir * 2;

  // ---- one-shot weight stage into LDS (wave w loads gates {2w, 2w+1}) ----
#pragma unroll
  for (int gg = 0; gg < 2; ++gg) {
    const int g = wave * 2 + gg;
    const __bf16* src = whh_d + (size_t)(g * H2_ + n0 + lr) * H2_;
    for (int kc = 0; kc < 16; ++kc) {
      const int k0 = kc * 32 + sel * 16;
      __bf16* dst = &wsm[((g * 16 + kc) * 32 + lane) * 16];
#if USE_ASYNC_LDS
      ASYNC_G2L_B128(src + k0, dst);
      ASYNC_G2L_B128(src + k0 + 8, dst + 8);
#else
      *reinterpret_cast<v8bf*>(dst)     = *reinterpret_cast<const v8bf*>(src + k0);
      *reinterpret_cast<v8bf*>(dst + 8) = *reinterpret_cast<const v8bf*>(src + k0 + 8);
#endif
    }
  }
#if USE_ASYNC_LDS
  __builtin_amdgcn_s_wait_asynccnt(0);
#endif
  __syncthreads();

  v8f c_state;
#pragma unroll
  for (int r = 0; r < 8; ++r) c_state[r] = 0.0f;

  for (int step = 0; step < T_; ++step) {
    const int t = (dir == 0) ? step : (T_ - 1 - step);
    const float* xg_t = xg_d + (size_t)t * XGSTEP_;

    // seed accumulators: one contiguous v8f per gate (fragment-major xg)
    v8f acc[4];
#pragma unroll
    for (int g = 0; g < 4; ++g) {
      acc[g] = *reinterpret_cast<const v8f*>(
          xg_t + (size_t)(((wave * 4 + g) * 32 + blk) * 32 + lane) * 8);
    }

    const __bf16* hprev = hb + (size_t)(step & 1) * B_ * H2_;
    const __bf16* abase = hprev + (size_t)m0 * H2_;
    for (int kc = 0; kc < 16; ++kc) {
      v16bf a = load_a_tile(abase, H2_, lr, sel, kc * 32);
#pragma unroll
      for (int g = 0; g < 4; ++g) {
        v16bf b = *reinterpret_cast<const v16bf*>(
            &wsm[((g * 16 + kc) * 32 + lane) * 16]);
        acc[g] = wmma_bf16(a, b, acc[g]);
      }
    }

    __bf16* hnext = hb + (size_t)((step + 1) & 1) * B_ * H2_;
    const int n = n0 + lr;
#pragma unroll
    for (int r = 0; r < 8; ++r) {
      int m = m0 + r + sel * 8;                  // batch index
      float gi = sigmoidf_(acc[0][r]);
      float gf = sigmoidf_(acc[1][r]);
      float gg = tanhf(acc[2][r]);
      float go = sigmoidf_(acc[3][r]);
      float c  = gf * c_state[r] + gi * gg;
      c_state[r] = c;
      float h = go * tanhf(c);
      __bf16 hb16 = (__bf16)h;
      hnext[(size_t)m * H2_ + n] = hb16;
      lstm_out[((size_t)m * T_ + t) * HD_ + dir * H2_ + n] = hb16;
    }

    grid_barrier(cnt, cnt + 1, 32u);
  }
}

// ---------------------------------------------------------------------------
// out[b,t,p] = 2*(x.p) - ||x||^2 - ||p||^2 ; grid = 1024 M-tiles x 8 waves(N)
// ---------------------------------------------------------------------------
__global__ void k_proto_out(const __bf16* __restrict__ lstm,
                            const __bf16* __restrict__ pro,
                            const float* __restrict__ x2,
                            const float* __restrict__ p2,
                            float* __restrict__ out) {
  const int M0   = blockIdx.x * 16;
  const int wave = threadIdx.x >> 5;
  const int lane = threadIdx.x & 31;
  const int lr = lane & 15, sel = lane >> 4;
  const int n0 = wave * 16;

  v8f acc;
#pragma unroll
  for (int r = 0; r < 8; ++r) acc[r] = 0.0f;

  const __bf16* abase = lstm + (size_t)M0 * HD_;
  for (int k0 = 0; k0 < HD_; k0 += 32) {
    v16bf a = load_a_tile(abase, HD_, lr, sel, k0);
    v16bf b = load_b_tile(pro, n0 + lr, HD_, sel, k0);
    acc = wmma_bf16(a, b, acc);
  }

  const int n = n0 + lr;
  const float pn = p2[n];
#pragma unroll
  for (int r = 0; r < 8; ++r) {
    int m = M0 + r + sel * 8;                    // global row b*T+t
    out[(size_t)m * P_ + n] = 2.0f * acc[r] - x2[m] - pn;
  }
}

// ---------------------------------------------------------------------------
// Host-side launcher
// ---------------------------------------------------------------------------
extern "C" void kernel_launch(void* const* d_in, const int* in_sizes, int n_in,
                              void* d_out, int out_size, void* d_ws, size_t ws_size,
                              hipStream_t stream) {
  (void)in_sizes; (void)n_in; (void)out_size; (void)ws_size;

  const int*   ids    = (const int*)d_in[0];
  const float* table  = (const float*)d_in[1];
  const float* wih_f  = (const float*)d_in[2];
  const float* whh_f  = (const float*)d_in[3];
  const float* bih_f  = (const float*)d_in[4];
  const float* bhh_f  = (const float*)d_in[5];
  const float* wih_b  = (const float*)d_in[6];
  const float* whh_b  = (const float*)d_in[7];
  const float* bih_b  = (const float*)d_in[8];
  const float* bhh_b  = (const float*)d_in[9];
  const float* protos = (const float*)d_in[10];
  float* out = (float*)d_out;

  char* ws = (char*)d_ws;
  __bf16*   xbf    = (__bf16*)(ws + OFF_XBF);
  __bf16*   wihbf  = (__bf16*)(ws + OFF_WIH);
  __bf16*   whhbf  = (__bf16*)(ws + OFF_WHH);
  __bf16*   probf  = (__bf16*)(ws + OFF_PRO);
  float*    bias   = (float*)(ws + OFF_BIAS);
  float*    p2     = (float*)(ws + OFF_P2);
  float*    xg     = (float*)(ws + OFF_XG);
  __bf16*   hbuf   = (__bf16*)(ws + OFF_HBUF);
  __bf16*   lstmbf = (__bf16*)(ws + OFF_LSTM);
  float*    x2     = (float*)(ws + OFF_X2);
  unsigned* bar    = (unsigned*)(ws + OFF_BAR);

  // init barriers + zero h double-buffers
  k_ws_init<<<256, 256, 0, stream>>>(hbuf, bar);

  // embedding gather -> bf16
  k_embed_bf<<<ROWS_, 256, 0, stream>>>(ids, table, xbf);

  // weight / prototype conversion
  const int NW = G4_ * E_;  // 1048576 per direction
  k_f2bf<<<(NW + 255) / 256, 256, 0, stream>>>(wih_f, wihbf, NW);
  k_f2bf<<<(NW + 255) / 256, 256, 0, stream>>>(wih_b, wihbf + NW, NW);
  k_f2bf<<<(NW + 255) / 256, 256, 0, stream>>>(whh_f, whhbf, NW);
  k_f2bf<<<(NW + 255) / 256, 256, 0, stream>>>(whh_b, whhbf + NW, NW);
  k_f2bf<<<(P_ * HD_ + 255) / 256, 256, 0, stream>>>(protos, probf, P_ * HD_);

  k_bias_sum<<<(G4_ + 255) / 256, 256, 0, stream>>>(bih_f, bhh_f, bih_b, bhh_b, bias);
  k_p2<<<P_, 256, 0, stream>>>(protos, p2);

  // input projection GEMM (both directions), fragment-major output
  dim3 gXG(G4_ / 128, ROWS_ / 64, 2);
  k_xg_gemm<<<gXG, 256, 0, stream>>>(xbf, wihbf, bias, xg);

  // persistent bidirectional recurrence (weights LDS-resident)
  k_lstm_recurrence<<<64, 64, 0, stream>>>(xg, whhbf, hbuf, lstmbf, bar);

  // ||x||^2 per (b,t) row
  k_row_sumsq<<<ROWS_, 256, 0, stream>>>(lstmbf, x2);

  // prototype distance output
  k_proto_out<<<ROWS_ / 16, 256, 0, stream>>>(lstmbf, probf, x2, p2, out);
}
